// SingleStrideAttention_28999619183179
// MI455X (gfx1250) — matrix operand order repaired and checked by reading, not compiled, not run on hardware
//
#include <hip/hip_runtime.h>
#include <hip/hip_bf16.h>
#include <math.h>

// ---------------- types ----------------
typedef __attribute__((ext_vector_type(16))) _Float16 v16h;
typedef __attribute__((ext_vector_type(8)))  _Float16 v8h;
typedef __attribute__((ext_vector_type(8)))  float    v8f;
typedef __attribute__((ext_vector_type(4)))  int      v4i;

#define D_MODEL 2048
#define GM      8192          // B*L = 2*4096
#define GN      2048
#define GK      2048
#define N_HEADS 16
#define DH      128
#define WIN     8
#define STRIDE_ 8
#define WELEMS  (2048*2048)

// ---------------- gfx1250 async global->LDS copy helpers ----------------
typedef __attribute__((address_space(1))) v4i* gv4i_p;   // global int4*
typedef __attribute__((address_space(3))) v4i* lv4i_p;   // LDS int4*

__device__ __forceinline__ void cp_async_b128(const void* g, void* l) {
    // GLOBAL_LOAD_ASYNC_TO_LDS_B128: tracked by ASYNCcnt, no VGPR staging
    __builtin_amdgcn_global_load_async_to_lds_b128(
        (gv4i_p)const_cast<void*>(g), (lv4i_p)l, 0, 0);
}
__device__ __forceinline__ void wait_async0() {
    __builtin_amdgcn_s_wait_asynccnt(0);
}

// ---------------- weight scale: two-stage deterministic reduction ----------------
__global__ __launch_bounds__(256) void wabs_partial(const float* __restrict__ w0,
                                                    const float* __restrict__ w1,
                                                    const float* __restrict__ w2,
                                                    const float* __restrict__ w3,
                                                    float* __restrict__ partial) {
    const float* w = (blockIdx.y == 0) ? w0 : (blockIdx.y == 1) ? w1
                   : (blockIdx.y == 2) ? w2 : w3;
    __shared__ float red[256];
    float s = 0.f;
    for (int i = blockIdx.x * 256 + threadIdx.x; i < WELEMS; i += 64 * 256)
        s += fabsf(w[i]);
    red[threadIdx.x] = s;
    __syncthreads();
    for (int o = 128; o > 0; o >>= 1) {
        if (threadIdx.x < o) red[threadIdx.x] += red[threadIdx.x + o];
        __syncthreads();
    }
    if (threadIdx.x == 0) partial[blockIdx.y * 64 + blockIdx.x] = red[0];
}

__global__ void wscale_final(const float* __restrict__ partial, float* __restrict__ scales) {
    if (threadIdx.x < 4) {
        float s = 0.f;
        for (int i = 0; i < 64; ++i) s += partial[threadIdx.x * 64 + i];
        s /= (float)WELEMS;
        scales[threadIdx.x] = fmaxf(s, 1e-5f);
    }
}

// ---------------- quantize weights to fp16 ternary ----------------
__global__ __launch_bounds__(256) void wquant(const float* __restrict__ w0,
                                              const float* __restrict__ w1,
                                              const float* __restrict__ w2,
                                              const float* __restrict__ w3,
                                              _Float16* __restrict__ q0,
                                              _Float16* __restrict__ q1,
                                              _Float16* __restrict__ q2,
                                              _Float16* __restrict__ q3,
                                              const float* __restrict__ scales) {
    const int mat = blockIdx.y;
    const float* w = (mat == 0) ? w0 : (mat == 1) ? w1 : (mat == 2) ? w2 : w3;
    _Float16*    q = (mat == 0) ? q0 : (mat == 1) ? q1 : (mat == 2) ? q2 : q3;
    const float s = scales[mat];
    const float inv = 1.0f / s;
    const int i = blockIdx.x * 256 + threadIdx.x;
    float t = rintf(fminf(fmaxf(w[i] * inv, -1.f), 1.f)) * s;
    q[i] = (_Float16)t;
}

// ---------------- RMSNorm + fp16 conversion (one block per row) ----------------
__global__ __launch_bounds__(256) void rms_convert(const float* __restrict__ x,
                                                   const float* __restrict__ g,
                                                   _Float16* __restrict__ xh,
                                                   _Float16* __restrict__ xnh) {
    const int row = blockIdx.x;
    const float* xr = x + (size_t)row * D_MODEL;
    __shared__ float red[256];
    float s = 0.f;
    float vbuf[8];
#pragma unroll
    for (int j = 0; j < 8; ++j) {
        float v = xr[threadIdx.x + j * 256];
        vbuf[j] = v;
        s += v * v;
    }
    red[threadIdx.x] = s;
    __syncthreads();
    for (int o = 128; o > 0; o >>= 1) {
        if (threadIdx.x < o) red[threadIdx.x] += red[threadIdx.x + o];
        __syncthreads();
    }
    const float r = rsqrtf(red[0] / (float)D_MODEL + 1e-6f);
#pragma unroll
    for (int j = 0; j < 8; ++j) {
        int c = threadIdx.x + j * 256;
        size_t off = (size_t)row * D_MODEL + c;
        xh[off]  = (_Float16)vbuf[j];
        xnh[off] = (_Float16)(vbuf[j] * r * g[c]);
    }
}

// ---------------- WMMA GEMM: Y[M,N] = A[M,K] * W[N,K]^T ----------------
// Block tile 128x128, 8 waves (2x4), wave tile 64x32 (4x2 WMMA frags).
// Double-buffered K-slab of 32 staged via GLOBAL_LOAD_ASYNC_TO_LDS_B128:
// issue slab s+1 async, compute WMMA on slab s, s_wait_asynccnt 0, barrier.
#define KSLAB 32
#define LDSS  40   // f16 stride (32 + 8 pad): 16B-aligned rows, bank-conflict-free

template <bool RES>
__global__ __launch_bounds__(256) void gemm_wmma(const _Float16* __restrict__ A,
                                                 const _Float16* __restrict__ W,
                                                 _Float16* __restrict__ outh,
                                                 float* __restrict__ outf,
                                                 const float* __restrict__ resid) {
    __shared__ __align__(16) _Float16 sA[2][128 * LDSS];
    __shared__ __align__(16) _Float16 sB[2][128 * LDSS];

    const int tid  = threadIdx.x;
    const int lane = tid & 31;
    const int wave = tid >> 5;
    const int wm   = wave >> 2;   // 0..1
    const int wn   = wave & 3;    // 0..3
    const int m0   = blockIdx.y * 128;
    const int n0   = blockIdx.x * 128;
    const int lr   = lane & 15;
    const int hs   = lane >> 4;   // lane half select

    // per-thread async-copy chunk coordinates: 512 chunks of 8 f16 per tile,
    // 2 chunks per thread per tile (A and B)
    const int r0 = tid >> 2;                // 0..63  (chunk set 0: rows 0..63)
    const int r1 = (tid >> 2) + 64;         // rows 64..127
    const int c0 = (tid & 3) << 3;          // 0,8,16,24

    v8f acc[4][2] = {};

    // ---- preload slab 0 ----
    cp_async_b128(&A[(size_t)(m0 + r0) * GK + c0], &sA[0][r0 * LDSS + c0]);
    cp_async_b128(&A[(size_t)(m0 + r1) * GK + c0], &sA[0][r1 * LDSS + c0]);
    cp_async_b128(&W[(size_t)(n0 + r0) * GK + c0], &sB[0][r0 * LDSS + c0]);
    cp_async_b128(&W[(size_t)(n0 + r1) * GK + c0], &sB[0][r1 * LDSS + c0]);
    wait_async0();
    __syncthreads();

    const int NS = GK / KSLAB;   // 64 slabs
    for (int s = 0; s < NS; ++s) {
        const int cur = s & 1;
        const int nxt = cur ^ 1;
        if (s + 1 < NS) {
            const int k0 = (s + 1) * KSLAB;
            cp_async_b128(&A[(size_t)(m0 + r0) * GK + k0 + c0], &sA[nxt][r0 * LDSS + c0]);
            cp_async_b128(&A[(size_t)(m0 + r1) * GK + k0 + c0], &sA[nxt][r1 * LDSS + c0]);
            cp_async_b128(&W[(size_t)(n0 + r0) * GK + k0 + c0], &sB[nxt][r0 * LDSS + c0]);
            cp_async_b128(&W[(size_t)(n0 + r1) * GK + k0 + c0], &sB[nxt][r1 * LDSS + c0]);
        }
        // A frags: lanes 0-15 hold K {0..7,16..23}, lanes 16-31 hold {8..15,24..31}
        v16h af[4];
#pragma unroll
        for (int mf = 0; mf < 4; ++mf) {
            const _Float16* p = &sA[cur][(wm * 64 + mf * 16 + lr) * LDSS + hs * 8];
            v8h lo = *(const v8h*)p;
            v8h hi = *(const v8h*)(p + 16);
            af[mf] = __builtin_shufflevector(lo, hi, 0, 1, 2, 3, 4, 5, 6, 7,
                                             8, 9, 10, 11, 12, 13, 14, 15);
        }
        // B frags: lanes 0-15 hold K {0..7,16..23}, lanes 16-31 hold {8..15,24..31}
        v16h bf[2];
#pragma unroll
        for (int nf = 0; nf < 2; ++nf) {
            const _Float16* p = &sB[cur][(wn * 32 + nf * 16 + lr) * LDSS + hs * 8];
            v8h lo = *(const v8h*)p;
            v8h hi = *(const v8h*)(p + 16);
            bf[nf] = __builtin_shufflevector(lo, hi, 0, 1, 2, 3, 4, 5, 6, 7,
                                             8, 9, 10, 11, 12, 13, 14, 15);
        }
#pragma unroll
        for (int mf = 0; mf < 4; ++mf)
#pragma unroll
            for (int nf = 0; nf < 2; ++nf)
                acc[mf][nf] = __builtin_amdgcn_wmma_f32_16x16x32_f16(
                    false, af[mf], false, bf[nf], (short)0, acc[mf][nf], false, false);

        wait_async0();
        __syncthreads();
    }

    // epilogue: D layout — VGPR v: M = v + 8*(lane/16), N = lane%16
#pragma unroll
    for (int mf = 0; mf < 4; ++mf) {
#pragma unroll
        for (int nf = 0; nf < 2; ++nf) {
            const int n  = n0 + wn * 32 + nf * 16 + lr;
            const int mb = m0 + wm * 64 + mf * 16 + hs * 8;
#pragma unroll
            for (int v = 0; v < 8; ++v) {
                size_t off = (size_t)(mb + v) * GN + n;
                if (RES) outf[off] = resid[off] + acc[mf][nf][v];
                else     outh[off] = (_Float16)acc[mf][nf][v];
            }
        }
    }
}

// ---------------- strided-window attention: one wave per (b,l,h) ----------------
__global__ __launch_bounds__(256) void attention_kernel(const _Float16* __restrict__ Qh,
                                                        const _Float16* __restrict__ Kh,
                                                        const _Float16* __restrict__ Vh,
                                                        _Float16* __restrict__ Oh) {
    const int item = blockIdx.x * 8 + (threadIdx.x >> 5);  // < B*L*H = 131072
    const int lane = threadIdx.x & 31;
    const int h  = item & (N_HEADS - 1);
    const int bl = item >> 4;            // b*L + l
    const int l  = bl & 4095;
    const float scale = 0.08838834764831845f;  // 128^-0.5

    const size_t base = (size_t)bl * D_MODEL + h * DH + lane * 4;
    const _Float16* qp = Qh + base;
    const float q0 = (float)qp[0], q1 = (float)qp[1], q2 = (float)qp[2], q3 = (float)qp[3];

    float logits[WIN];
#pragma unroll
    for (int w = 0; w < WIN; ++w) {
        const int pos = l - STRIDE_ * w;
        float part = 0.f;
        if (pos >= 0) {
            const _Float16* kp = Kh + (size_t)(bl - STRIDE_ * w) * D_MODEL + h * DH + lane * 4;
            part = q0 * (float)kp[0] + q1 * (float)kp[1] + q2 * (float)kp[2] + q3 * (float)kp[3];
        }
#pragma unroll
        for (int o = 16; o > 0; o >>= 1) part += __shfl_xor(part, o, 32);
        logits[w] = (pos >= 0) ? part * scale - 0.5f * logf((float)(STRIDE_ * w) + 1.0f)
                               : -INFINITY;
    }
    float mx = logits[0];
#pragma unroll
    for (int w = 1; w < WIN; ++w) mx = fmaxf(mx, logits[w]);
    float p[WIN], se = 0.f;
#pragma unroll
    for (int w = 0; w < WIN; ++w) { p[w] = expf(logits[w] - mx); se += p[w]; }
    const float inv = 1.0f / se;

    float a0 = 0.f, a1 = 0.f, a2 = 0.f, a3 = 0.f;
#pragma unroll
    for (int w = 0; w < WIN; ++w) {
        if (l - STRIDE_ * w >= 0) {
            const _Float16* vp = Vh + (size_t)(bl - STRIDE_ * w) * D_MODEL + h * DH + lane * 4;
            const float pw = p[w] * inv;
            a0 += pw * (float)vp[0];
            a1 += pw * (float)vp[1];
            a2 += pw * (float)vp[2];
            a3 += pw * (float)vp[3];
        }
    }
    _Float16* op = Oh + base;
    op[0] = (_Float16)a0; op[1] = (_Float16)a1; op[2] = (_Float16)a2; op[3] = (_Float16)a3;
}

// ---------------- launch ----------------
extern "C" void kernel_launch(void* const* d_in, const int* in_sizes, int n_in,
                              void* d_out, int out_size, void* d_ws, size_t ws_size,
                              hipStream_t stream) {
    (void)in_sizes; (void)n_in; (void)out_size; (void)ws_size;
    const float* x  = (const float*)d_in[0];
    const float* wq = (const float*)d_in[1];
    const float* wk = (const float*)d_in[2];
    const float* wv = (const float*)d_in[3];
    const float* wo = (const float*)d_in[4];
    const float* g  = (const float*)d_in[5];
    float* out = (float*)d_out;

    char* ws = (char*)d_ws;
    float* scales  = (float*)ws;
    float* partial = (float*)(ws + 1024);
    size_t off = 4096;
    const size_t ACT = (size_t)GM * D_MODEL * sizeof(_Float16);  // 33.5 MB
    const size_t WGT = (size_t)WELEMS * sizeof(_Float16);        //  8.4 MB
    _Float16* xh  = (_Float16*)(ws + off); off += ACT;
    _Float16* xnh = (_Float16*)(ws + off); off += ACT;
    _Float16* wqh = (_Float16*)(ws + off); off += WGT;
    _Float16* wkh = (_Float16*)(ws + off); off += WGT;
    _Float16* wvh = (_Float16*)(ws + off); off += WGT;
    _Float16* woh = (_Float16*)(ws + off); off += WGT;
    _Float16* Qh  = (_Float16*)(ws + off); off += ACT;
    _Float16* Kh  = (_Float16*)(ws + off); off += ACT;
    _Float16* Vh  = (_Float16*)(ws + off); off += ACT;
    _Float16* Oh  = (_Float16*)(ws + off); off += ACT;

    // 1) weight scales (deterministic two-stage)
    wabs_partial<<<dim3(64, 4), 256, 0, stream>>>(wq, wk, wv, wo, partial);
    wscale_final<<<1, 64, 0, stream>>>(partial, scales);
    // 2) ternary-quantize weights -> fp16
    wquant<<<dim3(WELEMS / 256, 4), 256, 0, stream>>>(wq, wk, wv, wo,
                                                      wqh, wkh, wvh, woh, scales);
    // 3) rmsnorm + fp16 conversion of activations
    rms_convert<<<GM, 256, 0, stream>>>(x, g, xh, xnh);
    // 4) Q/K/V projections (WMMA + async LDS pipeline)
    dim3 ggrid(GN / 128, GM / 128);
    gemm_wmma<false><<<ggrid, 256, 0, stream>>>(xnh, wqh, Qh, nullptr, nullptr);
    gemm_wmma<false><<<ggrid, 256, 0, stream>>>(xh,  wkh, Kh, nullptr, nullptr);
    gemm_wmma<false><<<ggrid, 256, 0, stream>>>(xh,  wvh, Vh, nullptr, nullptr);
    // 5) strided-window attention
    attention_kernel<<<(GM * N_HEADS) / 8, 256, 0, stream>>>(Qh, Kh, Vh, Oh);
    // 6) output projection + residual (WMMA, fp32 out)
    gemm_wmma<true><<<ggrid, 256, 0, stream>>>(Oh, woh, nullptr, out, x);
}
